// SignalCrossAttention_53343493816409
// MI455X (gfx1250) — matrix operand order repaired and checked by reading, not compiled, not run on hardware
//
#include <hip/hip_runtime.h>
#include <cstdint>
#include <cstddef>

typedef __attribute__((ext_vector_type(16))) _Float16 v16h;
typedef __attribute__((ext_vector_type(8)))  _Float16 v8h;
typedef __attribute__((ext_vector_type(8)))  float    v8f;

#define DEV __device__ __forceinline__

// ---------------------------------------------------------------------------
// WMMA helpers (CDNA5 16x16x32 f16 -> f32 accumulate)
// Fragment layouts per CDNA5 ISA 7.12.2:
//   A (16x32, MxK): lane l<16 -> row l, halves K{0..7} then K{16..23};
//                   lane l>=16 -> row l-16, halves K{8..15} then K{24..31}
//   B (32x16, KxN): lane l<16 -> col l, K{0..7} then K{16..23};
//                   lane l>=16 -> col l-16, K{8..15} then K{24..31}
//   C/D (16x16 f32): VGPR r: lanes 0-15 -> (M=r, N=lane); lanes 16-31 -> (M=r+8)
// ---------------------------------------------------------------------------
DEV v8f wmma32(v16h a, v16h b, v8f c) {
  return __builtin_amdgcn_wmma_f32_16x16x32_f16(false, a, false, b, (short)0, c,
                                                false, false);
}

DEV v16h load_a_frag(const _Float16* base, int ldh) {
  int lane = (int)(threadIdx.x & 31);
  const _Float16* p = base + (lane & 15) * ldh + ((lane >> 4) << 3);
  union { v16h v; v8h h[2]; } u;
  u.h[0] = *(const v8h*)(p);
  u.h[1] = *(const v8h*)(p + 16);
  return u.v;
}

DEV v16h load_b_frag(const _Float16* base, int ldh) {
  int lane = (int)(threadIdx.x & 31);
  const _Float16* p = base + (lane & 15) * ldh + ((lane >> 4) << 3);
  union { v16h v; v8h h[2]; } u;
  u.h[0] = *(const v8h*)(p);
  u.h[1] = *(const v8h*)(p + 16);
  return u.v;
}

// ---------------------------------------------------------------------------
// Async global->LDS copies (CDNA5 Tensor/Async path, tracked by ASYNCcnt).
// ISA 15.18.3: GLOBAL_LOAD_ASYNC_TO_LDS_B128; INST_OFFSET is added to BOTH the
// global address and the LDS address, so one (lds, gaddr) pair covers a 32B
// chunk with offsets 0/16. LDS aperture maps addr[31:0] directly to the
// wave-relative LDS byte address, so truncating the generic pointer is valid.
// ---------------------------------------------------------------------------
DEV void async_copy_32B(const _Float16* lds_dst, const _Float16* gsrc) {
  uint32_t l = (uint32_t)(size_t)(const void*)lds_dst;
  uint64_t g = (uint64_t)(size_t)(const void*)gsrc;
  asm volatile(
      "global_load_async_to_lds_b128 %0, %1, off\n\t"
      "global_load_async_to_lds_b128 %0, %1, off offset:16"
      :
      : "v"(l), "v"(g)
      : "memory");
}

DEV void async_copy_16B(const _Float16* lds_dst, const _Float16* gsrc) {
  uint32_t l = (uint32_t)(size_t)(const void*)lds_dst;
  uint64_t g = (uint64_t)(size_t)(const void*)gsrc;
  asm volatile("global_load_async_to_lds_b128 %0, %1, off"
               :
               : "v"(l), "v"(g)
               : "memory");
}

DEV void wait_async_le4() { asm volatile("s_wait_asynccnt 0x4" ::: "memory"); }
DEV void wait_async_0()   { asm volatile("s_wait_asynccnt 0x0" ::: "memory"); }

// ---------------------------------------------------------------------------
// fp32 -> fp16 conversion (weights)
// ---------------------------------------------------------------------------
__global__ __launch_bounds__(256) void f32_to_f16_kernel(
    const float* __restrict__ in, _Float16* __restrict__ out, int n) {
  int i = blockIdx.x * 256 + threadIdx.x;
  if (i < n) out[i] = (_Float16)in[i];
}

// ---------------------------------------------------------------------------
// Fused (optional embedding add) + LayerNorm. One row (D=512) per 256-thread
// block; wave32 shuffle reduction + cross-wave LDS combine.
// ---------------------------------------------------------------------------
template <bool EMB>
__global__ __launch_bounds__(256) void ln_kernel(
    const float* __restrict__ xin, const int* __restrict__ ids,
    const float* __restrict__ emb, const float* __restrict__ g,
    const float* __restrict__ bt, float* __restrict__ xres,
    _Float16* __restrict__ xn) {
  const int row = blockIdx.x;
  const int tid = threadIdx.x;
  const float* xr = xin + (size_t)row * 512;
  float a0 = xr[tid], a1 = xr[tid + 256];
  if (EMB) {
    int id = ids[row];
    a0 += emb[id * 512 + tid];
    a1 += emb[id * 512 + tid + 256];
    xres[(size_t)row * 512 + tid] = a0;
    xres[(size_t)row * 512 + tid + 256] = a1;
  }
  float s = a0 + a1, q = a0 * a0 + a1 * a1;
#pragma unroll
  for (int off = 16; off > 0; off >>= 1) {
    s += __shfl_xor(s, off);
    q += __shfl_xor(q, off);
  }
  __shared__ float ws[8], wq[8];
  __shared__ float s_mu, s_rs;
  if ((tid & 31) == 0) { ws[tid >> 5] = s; wq[tid >> 5] = q; }
  __syncthreads();
  if (tid == 0) {
    float S1 = 0.f, S2 = 0.f;
    for (int i = 0; i < 8; ++i) { S1 += ws[i]; S2 += wq[i]; }
    float mu = S1 * (1.0f / 512.0f);
    float var = S2 * (1.0f / 512.0f) - mu * mu;
    s_mu = mu;
    s_rs = rsqrtf(var + 1e-5f);
  }
  __syncthreads();
  float mu = s_mu, rs = s_rs;
  xn[(size_t)row * 512 + tid] =
      (_Float16)((a0 - mu) * rs * g[tid] + bt[tid]);
  xn[(size_t)row * 512 + tid + 256] =
      (_Float16)((a1 - mu) * rs * g[tid + 256] + bt[tid + 256]);
}

// ---------------------------------------------------------------------------
// WMMA NT-GEMM: C[M,N] = A[M,K] * W[N,K]^T + bias (+ epilogue)
// Block tile 128x128, BK=32, DOUBLE-BUFFERED LDS tiles filled with
// global_load_async_to_lds_b128: tile k+1 streams into LDS (ASYNCcnt) while
// 8 waves run 8 v_wmma_f32_16x16x32_f16 on tile k. Each thread issues 4 async
// b128s per tile; "s_wait_asynccnt 4" retires the current tile (in-order)
// while the next stays in flight.
// EPI: 0 = fp16 out; 1 = fp32 out + residual add; 2 = exact GELU, fp16 out.
// ---------------------------------------------------------------------------
template <int EPI>
__global__ __launch_bounds__(256) void gemm_nt(
    const _Float16* __restrict__ A, const _Float16* __restrict__ W,
    const float* __restrict__ bias, const float* __restrict__ res,
    float* __restrict__ outF, _Float16* __restrict__ outH, int M, int N,
    int K) {
  __shared__ _Float16 sA[2][128][40];  // +8 half pad vs LDS banks
  __shared__ _Float16 sB[2][128][40];
  const int tid = threadIdx.x;
  const int wave = tid >> 5;
  const int lane = tid & 31;
  const int wm = wave >> 2;  // 0..1
  const int wn = wave & 3;   // 0..3
  const int m0 = blockIdx.y * 128;
  const int n0 = blockIdx.x * 128;
  const int lrow = tid >> 1;        // 0..127
  const int lseg = (tid & 1) << 4;  // 0 or 16 halves

  const _Float16* gA = A + (size_t)(m0 + lrow) * K + lseg;
  const _Float16* gB = W + (size_t)(n0 + lrow) * K + lseg;

  v8f acc[4][2] = {};

  const int nk = K >> 5;
  // prologue: stage tile 0 into buffer 0
  async_copy_32B(&sA[0][lrow][lseg], gA);
  async_copy_32B(&sB[0][lrow][lseg], gB);

  for (int kt = 0; kt < nk; ++kt) {
    const int cur = kt & 1;
    if (kt + 1 < nk) {
      const int kof = (kt + 1) << 5;
      async_copy_32B(&sA[cur ^ 1][lrow][lseg], gA + kof);
      async_copy_32B(&sB[cur ^ 1][lrow][lseg], gB + kof);
      wait_async_le4();  // current tile (first 4 ops) has landed
    } else {
      wait_async_0();
    }
    __syncthreads();
    v16h bf0 = load_b_frag(&sB[cur][wn * 32][0], 40);
    v16h bf1 = load_b_frag(&sB[cur][wn * 32 + 16][0], 40);
#pragma unroll
    for (int i = 0; i < 4; ++i) {
      v16h af = load_a_frag(&sA[cur][wm * 64 + i * 16][0], 40);
      acc[i][0] = wmma32(af, bf0, acc[i][0]);
      acc[i][1] = wmma32(af, bf1, acc[i][1]);
    }
    __syncthreads();  // all waves done reading before buf is overwritten
  }

  const int rbase = m0 + wm * 64 + ((lane >> 4) << 3);
  const int cbase = n0 + wn * 32 + (lane & 15);
#pragma unroll
  for (int i = 0; i < 4; ++i) {
#pragma unroll
    for (int j = 0; j < 2; ++j) {
      const int col = cbase + j * 16;
      const float bz = bias[col];
#pragma unroll
      for (int r = 0; r < 8; ++r) {
        const int row = rbase + i * 16 + r;
        const size_t idx = (size_t)row * N + col;
        float v = acc[i][j][r] + bz;
        if (EPI == 0) {
          outH[idx] = (_Float16)v;
        } else if (EPI == 1) {
          outF[idx] = v + res[idx];
        } else {  // exact GELU: 0.5*x*(1+erf(x/sqrt(2)))
          outH[idx] = (_Float16)(0.5f * v *
                                 (1.0f + erff(v * 0.70710678118654752f)));
        }
      }
    }
  }
}

// ---------------------------------------------------------------------------
// Attention: block = (batch b, 32 query rows), 256 threads (8 waves), loops
// all 8 heads serially (exclusive ownership of attn-average rows ->
// deterministic, no atomics). Full 32x1024 score strip lives in LDS (128KB,
// feasible only on CDNA5's 320KB LDS), so softmax does NOT recompute QK^T.
// Q/K tiles stream in via global_load_async_to_lds_b128.
// qkv layout: [B*S][1536] fp16, cols 0..511 Q, 512..1023 K, 1024..1535 V,
// head h at offset h*64.
// ---------------------------------------------------------------------------
__global__ __launch_bounds__(256) void attn_kernel(
    const _Float16* __restrict__ qkv, _Float16* __restrict__ ctx,
    float* __restrict__ attn) {
  extern __shared__ char smem[];
  const int SCLD = 1032;  // padded row stride (floats) to break bank conflicts
  float* s_sc = (float*)smem;                   // 32 x 1032 f32
  _Float16* s_q = (_Float16*)(smem + 132096);   // 32 x 72 (Q, then P)
  _Float16* s_kv = (_Float16*)(smem + 136704);  // 64 x 72 (K, then V^T)
  float* s_max = (float*)(smem + 145920);       // 32
  float* s_sum = (float*)(smem + 146048);       // 32 (holds 1/sum)
  float* s_part = (float*)(smem + 146176);      // 32 x 8 partials

  const int b = blockIdx.x;           // 0..7
  const int qbase = blockIdx.y * 32;  // 0..992
  const int tid = threadIdx.x;
  const int wave = tid >> 5, lane = tid & 31;
  const int wm = wave >> 2, wn = wave & 3;  // 2x4 wave grid over 32x64 tiles

  for (int h = 0; h < 8; ++h) {
    __syncthreads();
    // ---- stage Q tile (32x64 halves) via async path ----
    {
      const int row = tid >> 3, seg = (tid & 7) << 3;
      const _Float16* src =
          qkv + (size_t)(b * 1024 + qbase + row) * 1536 + h * 64 + seg;
      async_copy_16B(&s_q[row * 72 + seg], src);
    }
    // ---- Phase A: all 1024 score columns into LDS ----
    for (int kb = 0; kb < 16; ++kb) {
      __syncthreads();
      {
        const int row = tid >> 2, seg = (tid & 3) << 4;
        const _Float16* src = qkv + (size_t)(b * 1024 + kb * 64 + row) * 1536 +
                              512 + h * 64 + seg;
        async_copy_32B(&s_kv[row * 72 + seg], src);
      }
      wait_async_0();
      __syncthreads();
      v8f c = {};
#pragma unroll
      for (int ks = 0; ks < 64; ks += 32) {
        v16h a = load_a_frag(s_q + wm * 16 * 72 + ks, 72);
        v16h bb = load_b_frag(s_kv + wn * 16 * 72 + ks, 72);
        c = wmma32(a, bb, c);
      }
      const int cr = wm * 16 + ((lane >> 4) << 3);
      const int cc = kb * 64 + wn * 16 + (lane & 15);
#pragma unroll
      for (int r = 0; r < 8; ++r)
        s_sc[(cr + r) * SCLD + cc] = c[r] * 0.125f;  // 1/sqrt(64)
    }
    __syncthreads();
    // ---- row max (8 partials per row) ----
    {
      const int rr = tid >> 3, pp = tid & 7;
      float m = -1e30f;
      for (int c2 = pp * 128; c2 < pp * 128 + 128; ++c2)
        m = fmaxf(m, s_sc[rr * SCLD + c2]);
      s_part[rr * 8 + pp] = m;
    }
    __syncthreads();
    if (tid < 32) {
      float m = -1e30f;
      for (int i = 0; i < 8; ++i) m = fmaxf(m, s_part[tid * 8 + i]);
      s_max[tid] = m;
    }
    __syncthreads();
    // ---- row sum of exp ----
    {
      const int rr = tid >> 3, pp = tid & 7;
      const float mf = s_max[rr];
      float s = 0.f;
      for (int c2 = pp * 128; c2 < pp * 128 + 128; ++c2)
        s += __expf(s_sc[rr * SCLD + c2] - mf);
      s_part[rr * 8 + pp] = s;
    }
    __syncthreads();
    if (tid < 32) {
      float s = 0.f;
      for (int i = 0; i < 8; ++i) s += s_part[tid * 8 + i];
      s_sum[tid] = 1.0f / s;
    }
    __syncthreads();

    // ---- Phase B: write averaged probs, accumulate ctx = P @ V ----
    v8f cctx = {};
    for (int kb = 0; kb < 16; ++kb) {
      __syncthreads();
      // V tile, stored transposed in LDS: s_kv[d][key] (scatter -> VGPR path)
      {
        const int krow = tid >> 2, seg = (tid & 3) << 4;
        const _Float16* src = qkv + (size_t)(b * 1024 + kb * 64 + krow) * 1536 +
                              1024 + h * 64 + seg;
        v8h v0 = *(const v8h*)(src);
        v8h v1 = *(const v8h*)(src + 8);
#pragma unroll
        for (int j = 0; j < 8; ++j) {
          s_kv[(seg + j) * 72 + krow] = v0[j];
          s_kv[(seg + 8 + j) * 72 + krow] = v1[j];
        }
      }
      // P tile (reuses Q buffer) + averaged attention output
      {
        const int row = tid >> 3, c0 = (tid & 7) << 3;
        const float mf = s_max[row], is = s_sum[row];
        float* ap =
            attn + (size_t)(b * 1024 + qbase + row) * 1024 + kb * 64 + c0;
#pragma unroll
        for (int j = 0; j < 8; ++j) {
          float p = __expf(s_sc[row * SCLD + kb * 64 + c0 + j] - mf) * is;
          s_q[row * 72 + c0 + j] = (_Float16)p;
          float contrib = p * 0.125f;  // mean over 8 heads
          if (h == 0) ap[j] = contrib; else ap[j] += contrib;
        }
      }
      __syncthreads();
#pragma unroll
      for (int ks = 0; ks < 64; ks += 32) {
        v16h a = load_a_frag(s_q + wm * 16 * 72 + ks, 72);
        v16h bb = load_b_frag(s_kv + wn * 16 * 72 + ks, 72);
        cctx = wmma32(a, bb, cctx);
      }
    }
    // ---- write ctx fragment: [B*S][512] fp16, col = h*64 + d ----
    {
      const int cr = qbase + wm * 16 + ((lane >> 4) << 3);
      const int cc = h * 64 + wn * 16 + (lane & 15);
#pragma unroll
      for (int r = 0; r < 8; ++r)
        ctx[(size_t)(b * 1024 + cr + r) * 512 + cc] = (_Float16)cctx[r];
    }
  }
}

// ---------------------------------------------------------------------------
// Host launch. Workspace layout (bytes), ~124 MB total:
//   xres1 f32 16MB | xres2 f32 16MB | xn f16 8MB | qkv f16 24MB | ctx f16 8MB
//   | h f16 8MB | h1 f16 32MB | wq f16 1.5MB | wo f16 0.5MB | w1 2MB | w2 2MB
// d_out: x [8,1024,512] f32 then attn [8,1024,1024] f32.
// ---------------------------------------------------------------------------
extern "C" void kernel_launch(void* const* d_in, const int* in_sizes, int n_in,
                              void* d_out, int out_size, void* d_ws,
                              size_t ws_size, hipStream_t stream) {
  (void)in_sizes; (void)n_in; (void)out_size; (void)ws_size;
  const float* x = (const float*)d_in[0];
  const int* ids = (const int*)d_in[1];
  const float* emb = (const float*)d_in[2];
  const float* in_w = (const float*)d_in[3];
  const float* in_b = (const float*)d_in[4];
  const float* out_w = (const float*)d_in[5];
  const float* out_b = (const float*)d_in[6];
  const float* ln1_g = (const float*)d_in[7];
  const float* ln1_b = (const float*)d_in[8];
  const float* ln2_g = (const float*)d_in[9];
  const float* ln2_b = (const float*)d_in[10];
  const float* w1 = (const float*)d_in[11];
  const float* b1 = (const float*)d_in[12];
  const float* w2 = (const float*)d_in[13];
  const float* b2 = (const float*)d_in[14];

  char* ws = (char*)d_ws;
  float* xres1 = (float*)(ws + 0);
  float* xres2 = (float*)(ws + 16777216);
  _Float16* xn = (_Float16*)(ws + 33554432);
  _Float16* qkvh = (_Float16*)(ws + 41943040);
  _Float16* ctxh = (_Float16*)(ws + 67108864);
  _Float16* hh = (_Float16*)(ws + 75497472);
  _Float16* h1h = (_Float16*)(ws + 83886080);
  _Float16* wq_h = (_Float16*)(ws + 117440512);
  _Float16* wo_h = (_Float16*)(ws + 119013376);
  _Float16* w1_h = (_Float16*)(ws + 119537664);
  _Float16* w2_h = (_Float16*)(ws + 121634816);

  float* out_x = (float*)d_out;
  float* out_attn = out_x + (size_t)8 * 1024 * 512;

  // weight fp16 copies
  f32_to_f16_kernel<<<(1536 * 512) / 256, 256, 0, stream>>>(in_w, wq_h,
                                                            1536 * 512);
  f32_to_f16_kernel<<<(512 * 512) / 256, 256, 0, stream>>>(out_w, wo_h,
                                                           512 * 512);
  f32_to_f16_kernel<<<(2048 * 512) / 256, 256, 0, stream>>>(w1, w1_h,
                                                            2048 * 512);
  f32_to_f16_kernel<<<(512 * 2048) / 256, 256, 0, stream>>>(w2, w2_h,
                                                            512 * 2048);

  // embed + LN1
  ln_kernel<true><<<8192, 256, 0, stream>>>(x, ids, emb, ln1_g, ln1_b, xres1,
                                            xn);

  // qkv = xn @ in_proj_w^T + b   [8192 x 1536]
  gemm_nt<0><<<dim3(12, 64), 256, 0, stream>>>(xn, wq_h, in_b, nullptr,
                                               nullptr, qkvh, 8192, 1536, 512);

  // attention (needs 147200 B dynamic LDS -> opt in above 64KB default)
  const int ATTN_SMEM = 147200;
  hipFuncSetAttribute((const void*)attn_kernel,
                      hipFuncAttributeMaxDynamicSharedMemorySize, ATTN_SMEM);
  attn_kernel<<<dim3(8, 32), 256, ATTN_SMEM, stream>>>(qkvh, ctxh, out_attn);

  // x = x + ctx @ out_proj_w^T + b   -> xres2
  gemm_nt<1><<<dim3(4, 64), 256, 0, stream>>>(ctxh, wo_h, out_b, xres1, xres2,
                                              nullptr, 8192, 512, 512);

  // LN2
  ln_kernel<false><<<8192, 256, 0, stream>>>(xres2, nullptr, nullptr, ln2_g,
                                             ln2_b, nullptr, hh);

  // h1 = gelu(h @ w1^T + b1)   [8192 x 2048]
  gemm_nt<2><<<dim3(16, 64), 256, 0, stream>>>(hh, w1_h, b1, nullptr, nullptr,
                                               h1h, 8192, 2048, 512);

  // out = xres2 + h1 @ w2^T + b2  -> d_out x-section
  gemm_nt<1><<<dim3(4, 64), 256, 0, stream>>>(h1h, w2_h, b2, xres2, out_x,
                                              nullptr, 8192, 512, 2048);
}
